// NAM_33947421507684
// MI455X (gfx1250) — matrix operand order repaired and checked by reading, not compiled, not run on hardware
//
#include <hip/hip_runtime.h>

// ---------------------------------------------------------------------------
// Fused bank-of-tiny-MLPs kernel for MI455X (gfx1250, wave32, WMMA).
//
// out[n,k] = beta[k] + sum_l alpha[l,k] * sum_p MLP_{k,l,p}(x[n,l])
//
// Features along WMMA M, samples along WMMA N. Each wave owns a 16-sample
// tile; two MLPs (a p-pair) are packed block-diagonally into the 16-wide
// feature dim, so one v_wmma_f32_16x16x32_f16 evaluates layer 2 (another
// layer 3) for 2 MLPs x 16 samples. Weight fragments are pre-packed into
// their exact f16 VGPR layout in LDS once per (k,l) by the whole block, so
// each wave fetches them with single ds_load_b128s.
// ---------------------------------------------------------------------------

typedef __attribute__((ext_vector_type(16))) _Float16 v16h;
typedef __attribute__((ext_vector_type(8)))  _Float16 v8h;
typedef __attribute__((ext_vector_type(8)))  float    v8f;
typedef __attribute__((ext_vector_type(4)))  float    v4f;
typedef __attribute__((ext_vector_type(4)))  unsigned int v4u;

#define N_TOT 32768
#define L_DIM 32
#define K_DIM 8
#define P_DIM 4
#define WAVES_PER_BLOCK 8   // 256 threads = 8 wave32

__device__ __forceinline__ unsigned int pack2h(float a, float b) {
    _Float16 ha = (_Float16)a, hb = (_Float16)b;
    return (unsigned int)__builtin_bit_cast(unsigned short, ha) |
           ((unsigned int)__builtin_bit_cast(unsigned short, hb) << 16);
}

__device__ __forceinline__ v8h relu8h(v8h v) {
#if __has_builtin(__builtin_elementwise_max)
    const v8h z = {};
    return __builtin_elementwise_max(v, z);
#else
    v8h r;
    #pragma unroll
    for (int i = 0; i < 8; ++i)
        r[i] = v[i] > (_Float16)0 ? v[i] : (_Float16)0;
    return r;
#endif
}

__global__ __launch_bounds__(256) void mlp_bank_kernel(
    const float* __restrict__ x,      // [N, L]
    const float* __restrict__ alpha,  // [L, K]
    const float* __restrict__ beta,   // [K]
    const float* __restrict__ W1,     // [K,L,P,8]
    const float* __restrict__ b1,     // [K,L,P,8]
    const float* __restrict__ W2,     // [K,L,P,8,8]
    const float* __restrict__ b2,     // [K,L,P,8]
    const float* __restrict__ W3,     // [K,L,P,8,6]
    const float* __restrict__ b3,     // [K,L,P,6]
    const float* __restrict__ W4,     // [K,L,P,6]
    const float* __restrict__ b4,     // [K,L,P]
    float* __restrict__ out)          // [N, K]
{
    // x tile for this block: [l][sample] to keep LDS reads conflict-free.
    __shared__ __align__(16) float sX[L_DIM * 128];            // 16 KB
    // Pre-packed f16 A-fragments in final VGPR layout: [pair][lane][4 dwords].
    __shared__ __align__(16) unsigned int sA2[2 * 32 * 4];     // 1 KB
    __shared__ __align__(16) unsigned int sA3[2 * 32 * 4];     // 1 KB
    // Layer-1 params as f16, plus a 32-half zero region read by all lanes as B elements 8..15.
    __shared__ __align__(16) _Float16 sW1h[64];
    __shared__ __align__(16) _Float16 sB1h[64];
    // Biases / layer-4 weights, padded to 8 floats per MLP.
    __shared__ __align__(16) float sB2f[P_DIM * 8];
    __shared__ __align__(16) float sB3f[P_DIM * 8];
    __shared__ __align__(16) float sW4f[P_DIM * 8];
    __shared__ float sB4[P_DIM];

    const int k    = blockIdx.y;
    const int tid  = threadIdx.x;
    const int wave = tid >> 5;
    const int lane = tid & 31;
    const int col  = lane & 15;   // sample within tile / feature column
    const int side = lane >> 4;   // 0: lanes 0-15, 1: lanes 16-31
    const int n0   = (blockIdx.x * WAVES_PER_BLOCK + wave) * 16;

    // ---- one-time staging: x tile + zero pads ---------------------------
    {
        const int blockN0 = blockIdx.x * 128;
        #pragma unroll
        for (int i = 0; i < 16; ++i) {
            const int idx = tid + i * 256;          // [s][l] order -> coalesced
            const int s = idx >> 5, ll = idx & 31;
            sX[ll * 128 + s] = x[(blockN0 + s) * L_DIM + ll];
        }
        if (tid < 32) { sW1h[32 + tid] = (_Float16)0; sB1h[32 + tid] = (_Float16)0; }
    }

    const float beta_k = beta[k];
    const v8h z8 = {};
    float acc = 0.0f;  // running sum over l and this lane's p

    // Staging-thread decomposition for fragment packing (one u32 each).
    const int s_pp   = tid >> 7;          // pair 0/1
    const int s_r    = tid & 127;
    const int s_lane = s_r >> 2;          // fragment lane 0..31
    const int s_d    = s_r & 3;           // dword 0..3 (halves 2d, 2d+1)
    const int s_col  = s_lane & 15;
    const int s_side = s_lane >> 4;
    const int s_pa   = 2 * s_pp, s_pb = 2 * s_pp + 1;

    for (int l = 0; l < L_DIM; ++l) {
        const int klp = (k * L_DIM + l) * P_DIM;

        __syncthreads();                       // prior readers done
        // ---- pack A2 = blockdiag(W2a^T, W2b^T) into VGPR layout ---------
        {
            const float* W2g = W2 + klp * 64;  // [p][h][g]
            const float* W3g = W3 + klp * 48;  // [p][h][g]
            const int e0 = 2 * s_d, e1 = 2 * s_d + 1;
            float a0 = 0.f, a1 = 0.f, c0 = 0.f, c1 = 0.f;
            if (s_side == 0) {
                if (s_col < 8) {
                    a0 = W2g[s_pa * 64 + e0 * 8 + s_col];
                    a1 = W2g[s_pa * 64 + e1 * 8 + s_col];
                }
                if (s_col < 6) {
                    c0 = W3g[s_pa * 48 + e0 * 6 + s_col];
                    c1 = W3g[s_pa * 48 + e1 * 6 + s_col];
                }
            } else {
                if (s_col >= 8) {
                    a0 = W2g[s_pb * 64 + e0 * 8 + (s_col - 8)];
                    a1 = W2g[s_pb * 64 + e1 * 8 + (s_col - 8)];
                    if (s_col < 14) {
                        c0 = W3g[s_pb * 48 + e0 * 6 + (s_col - 8)];
                        c1 = W3g[s_pb * 48 + e1 * 6 + (s_col - 8)];
                    }
                }
            }
            sA2[tid] = pack2h(a0, a1);
            sA3[tid] = pack2h(c0, c1);
        }
        if (tid < 32) {
            sW1h[tid] = (_Float16)W1[klp * 8 + tid];
            sB1h[tid] = (_Float16)b1[klp * 8 + tid];
            sB2f[tid] = b2[klp * 8 + tid];
            const int p = tid >> 3, j = tid & 7;
            sB3f[tid] = (j < 6) ? b3[klp * 6 + p * 6 + j] : 0.0f;
            sW4f[tid] = (j < 6) ? W4[klp * 6 + p * 6 + j] : 0.0f;
            if (tid < 4) sB4[tid] = b4[klp + tid];
        }
        __syncthreads();

        const float    xv       = sX[l * 128 + wave * 16 + col];
        const _Float16 xh       = (_Float16)xv;
        const v8h      xs       = {xh, xh, xh, xh, xh, xh, xh, xh};
        const float    alpha_lk = alpha[l * K_DIM + k];

        #pragma unroll
        for (int pp = 0; pp < 2; ++pp) {
            const int pa = 2 * pp, pb = 2 * pp + 1;
            const int ps = side ? pb : pa;

            // ---- A2 fragment: one ds_load_b128, upper half (K>=16) zero --
            const v4u ra2 = *reinterpret_cast<const v4u*>(&sA2[(pp * 32 + lane) * 4]);
            const v8h a2lo = __builtin_bit_cast(v8h, ra2);
            const v16h a2 = __builtin_shufflevector(a2lo, z8,
                0, 1, 2, 3, 4, 5, 6, 7, 8, 9, 10, 11, 12, 13, 14, 15);

            // ---- layer 1 in packed f16: all lanes read zeros for elements 8..15 -
            const int w1off_a = side ? pb * 8 : pa * 8;
            const int w1off_b = 32;
            const v8h w1a = *reinterpret_cast<const v8h*>(&sW1h[w1off_a]);
            const v8h c1a = *reinterpret_cast<const v8h*>(&sB1h[w1off_a]);
            const v8h w1b = *reinterpret_cast<const v8h*>(&sW1h[w1off_b]);
            const v8h c1b = *reinterpret_cast<const v8h*>(&sB1h[w1off_b]);
            const v8h ha = relu8h(w1a * xs + c1a);
            const v8h hb = relu8h(w1b * xs + c1b);
            const v16h b2f = __builtin_shufflevector(ha, hb,
                0, 1, 2, 3, 4, 5, 6, 7, 8, 9, 10, 11, 12, 13, 14, 15);

            // ---- C2 = layer-2 bias per feature row (two 16B LDS loads) ---
            const v4f* pb2 = reinterpret_cast<const v4f*>(&sB2f[ps * 8]);
            const v8f c2 = __builtin_shufflevector(pb2[0], pb2[1],
                                                   0, 1, 2, 3, 4, 5, 6, 7);

            const v8f d2 = __builtin_amdgcn_wmma_f32_16x16x32_f16(
                false, a2, false, b2f, (short)0, c2, false, false);

            // ---- B3 from relu(D2): each half keeps its own MLP's rows ----
            v16h b3f = {};
            #pragma unroll
            for (int j = 0; j < 8; ++j) {
                const float own = fmaxf(d2[j], 0.0f);                       // row 8*side+j
                b3f[j]     = (_Float16)own;   // elements 0..7 of this half; 8..15 = 0
            }

            // ---- A3 fragment ---------------------------------------------
            const v4u ra3 = *reinterpret_cast<const v4u*>(&sA3[(pp * 32 + lane) * 4]);
            const v8h a3lo = __builtin_bit_cast(v8h, ra3);
            const v16h a3 = __builtin_shufflevector(a3lo, z8,
                0, 1, 2, 3, 4, 5, 6, 7, 8, 9, 10, 11, 12, 13, 14, 15);

            const v4f* pb3 = reinterpret_cast<const v4f*>(&sB3f[ps * 8]);
            const v8f c3 = __builtin_shufflevector(pb3[0], pb3[1],
                                                   0, 1, 2, 3, 4, 5, 6, 7);

            const v8f d3 = __builtin_amdgcn_wmma_f32_16x16x32_f16(
                false, a3, false, b3f, (short)0, c3, false, false);

            // ---- layer 4: branch-free 8-wide dot (pads are exact zeros) --
            const v4f* pw4 = reinterpret_cast<const v4f*>(&sW4f[ps * 8]);
            const v8f w4v = __builtin_shufflevector(pw4[0], pw4[1],
                                                    0, 1, 2, 3, 4, 5, 6, 7);
            float o = sB4[ps];
            #pragma unroll
            for (int g = 0; g < 8; ++g)
                o = fmaf(w4v[g], fmaxf(d3[g], 0.0f), o);

            acc = fmaf(alpha_lk, o, acc);
        }
    }

    // combine the two p's held across wave halves, write out[n,k]
    const float tot = acc + __shfl_xor(acc, 16, 32);
    if (side == 0)
        out[(n0 + col) * K_DIM + k] = beta_k + tot;
}

extern "C" void kernel_launch(void* const* d_in, const int* in_sizes, int n_in,
                              void* d_out, int out_size, void* d_ws, size_t ws_size,
                              hipStream_t stream) {
    (void)in_sizes; (void)n_in; (void)d_ws; (void)ws_size; (void)out_size;
    const float* x     = (const float*)d_in[0];
    const float* alpha = (const float*)d_in[1];
    const float* beta  = (const float*)d_in[2];
    // d_in[3] = sigma (unused by the reference computation)
    const float* W1    = (const float*)d_in[4];
    const float* b1    = (const float*)d_in[5];
    const float* W2    = (const float*)d_in[6];
    const float* b2    = (const float*)d_in[7];
    const float* W3    = (const float*)d_in[8];
    const float* b3    = (const float*)d_in[9];
    const float* W4    = (const float*)d_in[10];
    const float* b4    = (const float*)d_in[11];
    float* out = (float*)d_out;

    dim3 grid(N_TOT / (16 * WAVES_PER_BLOCK), K_DIM);  // 256 x 8 blocks
    dim3 block(256);
    mlp_bank_kernel<<<grid, block, 0, stream>>>(
        x, alpha, beta, W1, b1, W2, b2, W3, b3, W4, b4, out);
}
